// EFLSTM_9629316678263
// MI455X (gfx1250) — matrix operand order repaired and checked by reading, not compiled, not run on hardware
//
#include <hip/hip_runtime.h>
#include <stdint.h>

typedef __attribute__((ext_vector_type(16))) __bf16 v16bf;
typedef __attribute__((ext_vector_type(8)))  float  v8f;

#define C_B   128
#define C_T   512
#define C_D   409
#define C_KX  416      /* D padded to multiple of 32 */
#define C_H   1024
#define C_G4  4096
#define C_C   512
#define C_O   7
#define NBLK_LSTM 32

// ---------- helpers ----------
__device__ __forceinline__ unsigned short f2bf(float f) {
  union { float f; unsigned u; } x; x.f = f;
  unsigned u = x.u;
  u += 0x7fffu + ((u >> 16) & 1u);           // round-to-nearest-even
  return (unsigned short)(u >> 16);
}
__device__ __forceinline__ float bf2f(unsigned short h) {
  union { unsigned u; float f; } x; x.u = ((unsigned)h) << 16;
  return x.f;
}
// A fragment (16x32 bf16, MxK). Lane L: M=L&15; lanes 0-15 hold K {0..7,16..23},
// lanes 16-31 hold K {8..15,24..31}. p points at (row, k0 + (lane>>4)*8).
__device__ __forceinline__ v16bf load_frag_a(const unsigned short* p) {
  union { v16bf v; uint4 q[2]; } u;
  u.q[0] = *(const uint4*)(p);
  u.q[1] = *(const uint4*)(p + 16);
  return u.v;
}
// B fragment (32x16 bf16, KxN). Lane L: N=L&15; K order same as A fragment.
// p points at (k0 + (lane>>4)*8, col), ld = row stride.
__device__ __forceinline__ v16bf load_frag_b(const unsigned short* p, int ld) {
  union { v16bf v; unsigned short s[16]; } u;
#pragma unroll
  for (int s = 0; s < 8; ++s) {
    u.s[s]     = p[(size_t)s * ld];
    u.s[8 + s] = p[(size_t)(16 + s) * ld];
  }
  return u.v;
}
__device__ __forceinline__ v8f wmma_bf16(v16bf a, v16bf b, v8f c) {
  return __builtin_amdgcn_wmma_f32_16x16x32_bf16(false, a, false, b, (short)0, c,
                                                 false, false);
}
__device__ __forceinline__ float sigmoidf_(float x) {
  return 1.0f / (1.0f + __expf(-x));
}

// ---------- packing / init kernels ----------
__global__ void pack_x_kernel(const float* __restrict__ mt, const float* __restrict__ ma,
                              const float* __restrict__ mv, unsigned short* __restrict__ Xp) {
  size_t idx = (size_t)blockIdx.x * 256 + threadIdx.x;
  if (idx >= (size_t)C_B * C_T * C_KX) return;
  int k = (int)(idx % C_KX);
  size_t row = idx / C_KX;                   // b*T + t
  float v = 0.f;
  if (k < 300)      v = mt[row * 300 + k];
  else if (k < 374) v = ma[row * 74 + (k - 300)];
  else if (k < C_D) v = mv[row * 35 + (k - 374)];
  Xp[idx] = f2bf(v);
}

// Wc[k][c] bf16, K = 1440 (rows 0..1023 = W_hh^T, 1024..1432 = W_ih^T, rest 0)
__global__ void pack_w_kernel(const float* __restrict__ Wih, const float* __restrict__ Whh,
                              unsigned short* __restrict__ Wc) {
  size_t idx = (size_t)blockIdx.x * 256 + threadIdx.x;
  if (idx >= (size_t)(C_H + C_KX) * C_G4) return;
  int c = (int)(idx % C_G4);
  int k = (int)(idx / C_G4);
  float v = 0.f;
  if (k < C_H)            v = Whh[(size_t)c * C_H + k];
  else if (k < C_H + C_D) v = Wih[(size_t)c * C_D + (k - C_H)];
  Wc[idx] = f2bf(v);
}

__global__ void pack_fc1t_kernel(const float* __restrict__ fc1w,
                                 unsigned short* __restrict__ fc1t) {
  size_t idx = (size_t)blockIdx.x * 256 + threadIdx.x;
  if (idx >= (size_t)C_H * C_C) return;
  int c = (int)(idx % C_C);
  int k = (int)(idx / C_C);
  fc1t[idx] = f2bf(fc1w[(size_t)c * C_H + k]);
}

__global__ void init_kernel(const float* __restrict__ bih, const float* __restrict__ bhh,
                            float* __restrict__ bias, unsigned short* __restrict__ hbuf0,
                            float* __restrict__ cstate, unsigned* __restrict__ ctl) {
  int idx = blockIdx.x * 256 + threadIdx.x;
  if (idx < C_G4) bias[idx] = bih[idx] + bhh[idx];
  if (idx < C_B * C_H) { hbuf0[idx] = 0; cstate[idx] = 0.f; }
  if (idx < 64) ctl[idx] = 0u;
}

// ---------- persistent LSTM recurrence ----------
// 32 WGs, WG ng owns hidden units [ng*32, ng*32+32) => 128 gate columns
// (4 strips of 32: i,f,g,o). M = full batch 128, K = 1440 = [h | x].
__global__ __launch_bounds__(256, 1) void lstm_persistent(
    const unsigned short* __restrict__ Wc, const float* __restrict__ bias,
    const unsigned short* __restrict__ Xp,
    unsigned short* hbuf0, unsigned short* hbuf1,
    float* __restrict__ cstate, unsigned short* __restrict__ h_all,
    unsigned* ctl) {
  __shared__ float gsh[64 * 128];            // 32 KB gate staging
  const int tid  = threadIdx.x;
  const int lane = tid & 31;
  const int wid  = tid >> 5;
  const int wm   = wid & 1;                  // 64-row half
  const int wsg  = wid >> 1;                 // gate strip 0..3
  const int mlo  = lane & 15;
  const int h8   = (lane >> 4) << 3;         // 0 or 8
  const int ng   = blockIdx.x;
  const int col0 = wsg * C_H + ng * 32;      // global gate column base

  unsigned* counter = ctl;
  unsigned* gen     = ctl + 1;

#pragma unroll 1
  for (int t = 0; t < C_T; ++t) {
    const unsigned short* hin  = (t & 1) ? hbuf1 : hbuf0;
    unsigned short*       hout = (t & 1) ? hbuf0 : hbuf1;

    // one speculative prefetch of the first weight rows for this step
    __builtin_prefetch(Wc + (size_t)lane * C_G4 + col0, 0, 1);

    v8f acc[4][2] = {};

    // recurrent part: h_t (128x1024) x W_hh^T
#pragma unroll 2
    for (int k0 = 0; k0 < C_H; k0 += 32) {
      const unsigned short* brow = Wc + (size_t)(k0 + h8) * C_G4 + col0 + mlo;
      v16bf b0 = load_frag_b(brow, C_G4);
      v16bf b1 = load_frag_b(brow + 16, C_G4);
#pragma unroll
      for (int i = 0; i < 4; ++i) {
        const int row = wm * 64 + i * 16 + mlo;
        v16bf a = load_frag_a(hin + (size_t)row * C_H + k0 + h8);
        acc[i][0] = wmma_bf16(a, b0, acc[i][0]);
        acc[i][1] = wmma_bf16(a, b1, acc[i][1]);
      }
    }
    // input part: x_t (128x416) x W_ih^T   (rows 1024.. of Wc)
#pragma unroll 2
    for (int k0 = 0; k0 < C_KX; k0 += 32) {
      const unsigned short* brow = Wc + (size_t)(C_H + k0 + h8) * C_G4 + col0 + mlo;
      v16bf b0 = load_frag_b(brow, C_G4);
      v16bf b1 = load_frag_b(brow + 16, C_G4);
#pragma unroll
      for (int i = 0; i < 4; ++i) {
        const int row = wm * 64 + i * 16 + mlo;
        v16bf a = load_frag_a(Xp + ((size_t)row * C_T + t) * C_KX + k0 + h8);
        acc[i][0] = wmma_bf16(a, b0, acc[i][0]);
        acc[i][1] = wmma_bf16(a, b1, acc[i][1]);
      }
    }

    // epilogue: two 64-row phases through LDS, then LSTM elementwise
#pragma unroll 1
    for (int ph = 0; ph < 2; ++ph) {
      __syncthreads();
      if (wm == ph) {
#pragma unroll
        for (int i = 0; i < 4; ++i)
#pragma unroll
          for (int n = 0; n < 2; ++n)
#pragma unroll
            for (int r = 0; r < 8; ++r) {
              int rr = i * 16 + h8 + r;              // C-layout: row = half*8 + r
              int cc = wsg * 32 + n * 16 + mlo;      // col = lane&15
              gsh[rr * 128 + cc] = acc[i][n][r];
            }
      }
      __syncthreads();
#pragma unroll 1
      for (int e = tid; e < 64 * 32; e += 256) {
        int row  = e >> 5;
        int u    = e & 31;
        int b    = ph * 64 + row;
        int unit = ng * 32 + u;
        float gi = gsh[row * 128 +       u] + bias[unit];
        float gf = gsh[row * 128 + 32  + u] + bias[C_H + unit];
        float gg = gsh[row * 128 + 64  + u] + bias[2 * C_H + unit];
        float go = gsh[row * 128 + 96  + u] + bias[3 * C_H + unit];
        float iv = sigmoidf_(gi);
        float fv = sigmoidf_(gf);
        float gv = tanhf(gg);
        float ov = sigmoidf_(go);
        size_t ci = (size_t)b * C_H + unit;
        float cn = fv * cstate[ci] + iv * gv;
        cstate[ci] = cn;
        unsigned short hb = f2bf(ov * tanhf(cn));
        hout[ci] = hb;
        h_all[((size_t)b * C_T + t) * C_H + unit] = hb;
      }
    }

    // device-wide barrier: publish h_{t+1}
    __builtin_amdgcn_fence(__ATOMIC_RELEASE, "agent");
    __syncthreads();
    if (tid == 0) {
      unsigned arr  = __hip_atomic_fetch_add(counter, 1u, __ATOMIC_ACQ_REL,
                                             __HIP_MEMORY_SCOPE_AGENT);
      unsigned need = arr / NBLK_LSTM + 1u;
      if ((arr + 1u) % NBLK_LSTM == 0u) {
        __hip_atomic_fetch_add(gen, 1u, __ATOMIC_ACQ_REL, __HIP_MEMORY_SCOPE_AGENT);
      } else {
        while (__hip_atomic_load(gen, __ATOMIC_ACQUIRE, __HIP_MEMORY_SCOPE_AGENT) < need)
          __builtin_amdgcn_s_sleep(2);
      }
    }
    __syncthreads();
    __builtin_amdgcn_fence(__ATOMIC_ACQUIRE, "agent");
  }
}

// ---------- head: hidden = relu(h_all x fc1^T + b1), bf16 WMMA ----------
__global__ __launch_bounds__(256, 1) void head1_gemm(
    const unsigned short* __restrict__ h_all, const unsigned short* __restrict__ fc1t,
    const float* __restrict__ fc1b, unsigned short* __restrict__ hidden) {
  const int tid  = threadIdx.x;
  const int lane = tid & 31;
  const int wid  = tid >> 5;
  const int wm   = wid & 1;
  const int wn   = wid >> 1;
  const int mlo  = lane & 15;
  const int h8   = (lane >> 4) << 3;
  const int m0   = blockIdx.x * 128;
  const int n0   = blockIdx.y * 128;

  v8f acc[4][2] = {};
#pragma unroll 2
  for (int k0 = 0; k0 < C_H; k0 += 32) {
    const unsigned short* brow = fc1t + (size_t)(k0 + h8) * C_C + n0 + wn * 32 + mlo;
    v16bf b0 = load_frag_b(brow, C_C);
    v16bf b1 = load_frag_b(brow + 16, C_C);
#pragma unroll
    for (int i = 0; i < 4; ++i) {
      const int row = m0 + wm * 64 + i * 16 + mlo;
      v16bf a = load_frag_a(h_all + (size_t)row * C_H + k0 + h8);
      acc[i][0] = wmma_bf16(a, b0, acc[i][0]);
      acc[i][1] = wmma_bf16(a, b1, acc[i][1]);
    }
  }
#pragma unroll
  for (int i = 0; i < 4; ++i)
#pragma unroll
    for (int n = 0; n < 2; ++n)
#pragma unroll
      for (int r = 0; r < 8; ++r) {
        int grow = m0 + wm * 64 + i * 16 + h8 + r;
        int gcol = n0 + wn * 32 + n * 16 + mlo;
        float v = acc[i][n][r] + fc1b[gcol];
        hidden[(size_t)grow * C_C + gcol] = f2bf(fmaxf(v, 0.0f));
      }
}

// ---------- head2: out = log_softmax(hidden x fc2^T + b2) ----------
__global__ __launch_bounds__(256) void head2_out(
    const unsigned short* __restrict__ hidden, const float* __restrict__ fc2w,
    const float* __restrict__ fc2b, float* __restrict__ out) {
  __shared__ float w[C_O * C_C];             // 14 KB
  const int tid = threadIdx.x;
  for (int i = tid; i < C_O * C_C; i += 256) w[i] = fc2w[i];
  __syncthreads();
  const int row = blockIdx.x * 256 + tid;    // 0..65535
  float a[C_O];
#pragma unroll
  for (int o = 0; o < C_O; ++o) a[o] = fc2b[o];
  const unsigned short* hr = hidden + (size_t)row * C_C;
#pragma unroll 4
  for (int k = 0; k < C_C; ++k) {
    float hv = bf2f(hr[k]);
#pragma unroll
    for (int o = 0; o < C_O; ++o) a[o] += hv * w[o * C_C + k];
  }
  float m = a[0];
#pragma unroll
  for (int o = 1; o < C_O; ++o) m = fmaxf(m, a[o]);
  float s = 0.f;
#pragma unroll
  for (int o = 0; o < C_O; ++o) s += __expf(a[o] - m);
  float l = m + __logf(s);
#pragma unroll
  for (int o = 0; o < C_O; ++o) out[(size_t)row * C_O + o] = a[o] - l;
}

extern "C" void kernel_launch(void* const* d_in, const int* in_sizes, int n_in,
                              void* d_out, int out_size, void* d_ws, size_t ws_size,
                              hipStream_t stream) {
  (void)in_sizes; (void)n_in; (void)out_size; (void)ws_size;
  const float* m_text  = (const float*)d_in[0];
  const float* m_audio = (const float*)d_in[1];
  const float* m_video = (const float*)d_in[2];
  const float* W_ih  = (const float*)d_in[5];
  const float* W_hh  = (const float*)d_in[6];
  const float* b_ih  = (const float*)d_in[7];
  const float* b_hh  = (const float*)d_in[8];
  const float* fc1_w = (const float*)d_in[9];
  const float* fc1_b = (const float*)d_in[10];
  const float* fc2_w = (const float*)d_in[11];
  const float* fc2_b = (const float*)d_in[12];
  float* out = (float*)d_out;

  char* ws = (char*)d_ws;
  unsigned short* Xp     = (unsigned short*)ws; ws += (size_t)C_B * C_T * C_KX * 2;
  unsigned short* Wc     = (unsigned short*)ws; ws += (size_t)(C_H + C_KX) * C_G4 * 2;
  unsigned short* fc1t   = (unsigned short*)ws; ws += (size_t)C_H * C_C * 2;
  unsigned short* hall   = (unsigned short*)ws; ws += (size_t)C_B * C_T * C_H * 2;
  unsigned short* hidden = (unsigned short*)ws; ws += (size_t)C_B * C_T * C_C * 2;
  unsigned short* hbuf0  = (unsigned short*)ws; ws += (size_t)C_B * C_H * 2;
  unsigned short* hbuf1  = (unsigned short*)ws; ws += (size_t)C_B * C_H * 2;
  float*    bias = (float*)ws;                  ws += (size_t)C_G4 * 4;
  float*    cst  = (float*)ws;                  ws += (size_t)C_B * C_H * 4;
  unsigned* ctl  = (unsigned*)ws;               ws += 256;

  pack_x_kernel<<<(unsigned)(((size_t)C_B * C_T * C_KX + 255) / 256), 256, 0, stream>>>(
      m_text, m_audio, m_video, Xp);
  pack_w_kernel<<<(unsigned)(((size_t)(C_H + C_KX) * C_G4 + 255) / 256), 256, 0, stream>>>(
      W_ih, W_hh, Wc);
  pack_fc1t_kernel<<<(unsigned)(((size_t)C_H * C_C + 255) / 256), 256, 0, stream>>>(
      fc1_w, fc1t);
  init_kernel<<<(C_B * C_H + 255) / 256, 256, 0, stream>>>(b_ih, b_hh, bias, hbuf0, cst, ctl);
  lstm_persistent<<<NBLK_LSTM, 256, 0, stream>>>(Wc, bias, Xp, hbuf0, hbuf1, cst, hall, ctl);
  head1_gemm<<<dim3((C_B * C_T) / 128, C_C / 128), 256, 0, stream>>>(hall, fc1t, fc1_b, hidden);
  head2_out<<<(C_B * C_T) / 256, 256, 0, stream>>>(hidden, fc2_w, fc2_b, out);
}
